// MultiHeadDotProductAttention_28604482191570
// MI455X (gfx1250) — matrix operand order repaired and checked by reading, not compiled, not run on hardware
//
#include <hip/hip_runtime.h>

typedef _Float16 v4h  __attribute__((ext_vector_type(4)));
typedef _Float16 v8h  __attribute__((ext_vector_type(8)));
typedef _Float16 v16h __attribute__((ext_vector_type(16)));
typedef float    v8f  __attribute__((ext_vector_type(8)));

// ---------------------------------------------------------------------------
// CDNA5 async global->LDS copy (ASYNCcnt-tracked DMA, 16 bytes per lane)
// ---------------------------------------------------------------------------
__device__ __forceinline__ unsigned int lds_off(const void* p) {
  // generic LDS address: [31:0] is the LDS byte offset (ISA aperture rule)
  return (unsigned int)(unsigned long long)p;
}
__device__ __forceinline__ void async_ld16(unsigned int lds, const void* gptr) {
  asm volatile("global_load_async_to_lds_b128 %0, %1, off"
               :
               : "v"(lds), "v"((unsigned long long)gptr)
               : "memory");
}
__device__ __forceinline__ void wait_async0() {
  asm volatile("s_wait_asynccnt 0" ::: "memory");
}

// ---------------------------------------------------------------------------
// CDNA5 Tensor Data Mover: 2-D tile DMA global->LDS (TENSORcnt-tracked)
// ---------------------------------------------------------------------------
#if __has_builtin(__builtin_amdgcn_tensor_load_to_lds)
#define HAVE_TDM 1
#else
#define HAVE_TDM 0
#endif

#if HAVE_TDM
typedef unsigned int v4u __attribute__((ext_vector_type(4)));
typedef int          v8i __attribute__((ext_vector_type(8)));
typedef int          v4i __attribute__((ext_vector_type(4)));

// D# per CDNA5 ISA: group0 = flags/lds/global addr,
// group1 = dims/tile/strides (2-byte elements), groups 2/3 unused (2-D tile).
__device__ __forceinline__ void tdm_load_2d(
    unsigned int lds_byte, const void* gptr,
    unsigned int tile_d0, unsigned int tile_d1,
    unsigned long long tensor_d0, unsigned long long tensor_d1,
    unsigned long long stride0_elems) {
  const unsigned long long ga = (unsigned long long)gptr;
  v4u g0 = { 1u,                                   // count=1 valid descriptor
             lds_byte,                             // LDS byte address
             (unsigned int)(ga & 0xFFFFFFFFu),     // global_addr[31:0]
             (unsigned int)(((ga >> 32) & 0x01FFFFFFu) | (2u << 30)) }; // +type=2
  v8i g1 = { (int)(1u << 16),                      // data_size = 1 (2 bytes)
             (int)((tensor_d0 & 0xFFFFu) << 16),   // tensor_dim0[15:0]
             (int)(((tensor_d0 >> 16) & 0xFFFFu) | ((tensor_d1 & 0xFFFFu) << 16)),
             (int)(((tensor_d1 >> 16) & 0xFFFFu) | (tile_d0 << 16)),
             (int)(tile_d1 & 0xFFFFu),             // tile_dim1, tile_dim2=0
             (int)(stride0_elems & 0xFFFFFFFFu),   // tensor_dim0_stride[31:0]
             (int)((stride0_elems >> 32) & 0xFFFFu),
             0 };
  v4i g2 = { 0, 0, 0, 0 };
  v4i g3 = { 0, 0, 0, 0 };
#if __has_include(<hip/amd_detail/amd_gfx1250_TDM.h>)
  v8i g4 = { 0, 0, 0, 0, 0, 0, 0, 0 };
  __builtin_amdgcn_tensor_load_to_lds(g0, g1, g2, g3, g4, 0);
#else
  __builtin_amdgcn_tensor_load_to_lds(g0, g1, g2, g3, 0);
#endif
}
__device__ __forceinline__ void wait_tensor0() {
#if __has_builtin(__builtin_amdgcn_s_wait_tensorcnt)
  __builtin_amdgcn_s_wait_tensorcnt(0);
#else
  asm volatile("s_wait_tensorcnt 0" ::: "memory");
#endif
}
#endif  // HAVE_TDM

// ---------------------------------------------------------------------------
// WMMA helpers (CDNA5 wave32, 16x16x32 f16 -> f32 accum)
// ---------------------------------------------------------------------------
__device__ __forceinline__ v8f wmma32(v16h a, v16h b, v8f c) {
  return __builtin_amdgcn_wmma_f32_16x16x32_f16(
      /*neg_a=*/false, a, /*neg_b=*/false, b,
      /*c_mod=*/(short)0, c, /*reuse_a=*/false, /*reuse_b=*/false);
}

// A-matrix 16x32 fragment.  ISA layout: lanes 0-15 hold row M=lane with
// K = {0..7, 16..23}; lanes 16-31 hold row M=lane-16 with K = {8..15, 24..31}.
__device__ __forceinline__ v16h frag_a(const _Float16* base, int ld, int k0) {
  const int lane = threadIdx.x & 31;
  const _Float16* p = base + (lane & 15) * ld + k0 + ((lane >> 4) << 3);
  v8h lo = *(const v8h*)p;
  v8h hi = *(const v8h*)(p + 16);
  return __builtin_shufflevector(lo, hi, 0, 1, 2, 3, 4, 5, 6, 7,
                                 8, 9, 10, 11, 12, 13, 14, 15);
}

// B-matrix 32x16 fragment from an [N][K] (K-contiguous) tile.
// Lanes 0-15: col N=lane, K={0..7,16..23}; lanes 16-31: col N=lane-16, K={8..15,24..31}.
__device__ __forceinline__ v16h frag_b(const _Float16* base, int ld, int k0) {
  const int lane = threadIdx.x & 31;
  const _Float16* p = base + (lane & 15) * ld + k0 + ((lane >> 4) << 3);
  v8h lo = *(const v8h*)p;
  v8h hi = *(const v8h*)(p + 16);
  return __builtin_shufflevector(lo, hi, 0, 1, 2, 3, 4, 5, 6, 7,
                                 8, 9, 10, 11, 12, 13, 14, 15);
}

// ---------------------------------------------------------------------------
// f32 -> f16 conversion (vectorized)
// ---------------------------------------------------------------------------
__global__ __launch_bounds__(256) void cvt4_f32_f16(
    const float* __restrict__ in, _Float16* __restrict__ out, int n4) {
  int i = blockIdx.x * 256 + threadIdx.x;
  if (i >= n4) return;
  float4 f = ((const float4*)in)[i];
  v4h h = { (_Float16)f.x, (_Float16)f.y, (_Float16)f.z, (_Float16)f.w };
  ((v4h*)out)[i] = h;
}

// Transpose + convert: in f32 [rows][cols] -> out f16 [cols][rows]
__global__ __launch_bounds__(256) void transpose_cvt(
    const float* __restrict__ in, _Float16* __restrict__ out, int rows, int cols) {
  __shared__ float tile[32][33];
  const int bx = blockIdx.x * 32;  // col base
  const int by = blockIdx.y * 32;  // row base
  const int tx = threadIdx.x, ty = threadIdx.y;
#pragma unroll
  for (int j = 0; j < 32; j += 8)
    tile[ty + j][tx] = in[(size_t)(by + ty + j) * cols + (bx + tx)];
  __syncthreads();
#pragma unroll
  for (int j = 0; j < 32; j += 8)
    out[(size_t)(bx + ty + j) * rows + (by + tx)] = (_Float16)tile[tx][ty + j];
}

// ---------------------------------------------------------------------------
// Tiled f16 GEMM:  C[M][N] = scale * A[M][K] * Bt[N][K]^T
// 128x128 workgroup tile, 8 waves (4 in M x 2 in N), wave tile 32x64.
// Double-buffered LDS; tiles filled by the Tensor Data Mover (one 2-D
// descriptor per tile, issued by wave 0) so the global->LDS traffic runs on
// the DMA engine while all waves stay in the WMMA pipe.  Falls back to
// per-lane async-to-LDS copies if the TDM builtin is unavailable.
// ---------------------------------------------------------------------------
template <bool OUTF32>
__global__ __launch_bounds__(256) void gemm_h(
    const _Float16* __restrict__ A, const _Float16* __restrict__ Bt,
    void* __restrict__ Cout, int M, int N, int K, float scale) {
  __shared__ __align__(16) _Float16 sA[2][128 * 32];
  __shared__ __align__(16) _Float16 sB[2][128 * 32];
  const int tid = threadIdx.x;
  const int lane = tid & 31, wave = tid >> 5;
  const int wm = wave & 3, wn = wave >> 2;
  const int m0 = blockIdx.y * 128, n0 = blockIdx.x * 128;

  v8f acc[2][4];
#pragma unroll
  for (int i = 0; i < 2; ++i)
#pragma unroll
    for (int j = 0; j < 4; ++j) { v8f z = {}; acc[i][j] = z; }

  const int row = tid >> 1;          // 0..127
  const int koff = (tid & 1) << 4;   // 0 or 16

  auto issue_tile = [&](int buf, int kk) {
#if HAVE_TDM
    if (wave == 0) {
      // A tile: 128 rows x 32 halfs, row stride K -> LDS [row][32] packed
      tdm_load_2d(lds_off(&sA[buf][0]), A + (size_t)m0 * K + kk,
                  /*tile_d0=*/32, /*tile_d1=*/128,
                  (unsigned long long)K, (unsigned long long)M,
                  (unsigned long long)K);
      // B tile: 128 cols x 32 halfs
      tdm_load_2d(lds_off(&sB[buf][0]), Bt + (size_t)n0 * K + kk,
                  /*tile_d0=*/32, /*tile_d1=*/128,
                  (unsigned long long)K, (unsigned long long)N,
                  (unsigned long long)K);
    }
#else
    const _Float16* ga = A + (size_t)(m0 + row) * K + kk + koff;
    async_ld16(lds_off(&sA[buf][row * 32 + koff]), ga);
    async_ld16(lds_off(&sA[buf][row * 32 + koff + 8]), ga + 8);
    const _Float16* gb = Bt + (size_t)(n0 + row) * K + kk + koff;
    async_ld16(lds_off(&sB[buf][row * 32 + koff]), gb);
    async_ld16(lds_off(&sB[buf][row * 32 + koff + 8]), gb + 8);
#endif
  };
  auto wait_tiles = [&]() {
#if HAVE_TDM
    if (wave == 0) wait_tensor0();
#else
    wait_async0();
#endif
  };

  // prologue: fill buffer 0
  issue_tile(0, 0);
  wait_tiles();
  __syncthreads();

  int cur = 0;
  for (int kk = 0; kk < K; kk += 32) {
    if (kk + 32 < K) issue_tile(cur ^ 1, kk + 32);  // DMA next slab

    v16h af[2], bf[4];
#pragma unroll
    for (int mt = 0; mt < 2; ++mt)
      af[mt] = frag_a(&sA[cur][(wm * 32 + mt * 16) * 32], 32, 0);
#pragma unroll
    for (int nt = 0; nt < 4; ++nt)
      bf[nt] = frag_b(&sB[cur][(wn * 64 + nt * 16) * 32], 32, 0);
#pragma unroll
    for (int mt = 0; mt < 2; ++mt)
#pragma unroll
      for (int nt = 0; nt < 4; ++nt)
        acc[mt][nt] = wmma32(af[mt], bf[nt], acc[mt][nt]);

    wait_tiles();       // next slab landed in LDS
    __syncthreads();    // one barrier per K-step
    cur ^= 1;
  }

  // C layout: VGPR i, lanes 0-15 -> (M=i, N=lane); lanes 16-31 -> (M=i+8).
  const int rsel = (lane >> 4) * 8;
  const int csel = lane & 15;
#pragma unroll
  for (int mt = 0; mt < 2; ++mt)
#pragma unroll
    for (int nt = 0; nt < 4; ++nt)
#pragma unroll
      for (int i = 0; i < 8; ++i) {
        const int r = m0 + wm * 32 + mt * 16 + i + rsel;
        const int c = n0 + wn * 64 + nt * 16 + csel;
        const float v = acc[mt][nt][i] * scale;
        if constexpr (OUTF32)
          ((float*)Cout)[(size_t)r * N + c] = v;
        else
          ((_Float16*)Cout)[(size_t)r * N + c] = (_Float16)v;
      }
}

// ---------------------------------------------------------------------------
// Flash attention.  Grid: (S/128, B*H).  Block: 256 (8 waves, 16 q-rows each).
// Q/K/V/X layout: [B,S,H,HD] f16 (row stride H*HD).  Q pre-scaled by 1/sqrt(HD).
// K tile filled by async global->LDS DMA; V transposed through VGPRs.
// ---------------------------------------------------------------------------
__global__ __launch_bounds__(256) void attn_h(
    const _Float16* __restrict__ Q, const _Float16* __restrict__ Km,
    const _Float16* __restrict__ Vm, const int* __restrict__ mask,
    _Float16* __restrict__ X) {
  constexpr int S = 2048, HD = 128, KB = 64, ROWSTR = 2048;
  __shared__ __align__(16) _Float16 sK[KB * HD];       // [kv][hd]   16 KB
  __shared__ __align__(16) _Float16 sVt[HD * KB];      // [hd][kv]   16 KB
  __shared__ __align__(16) _Float16 sP[8 * 16 * KB];   // per-wave P 16 KB

  const int tid = threadIdx.x, lane = tid & 31, wave = tid >> 5;
  const int b = blockIdx.y >> 4, h = blockIdx.y & 15;
  const int q0 = blockIdx.x * 128;
  const int qr0 = q0 + wave * 16;
  const size_t bh = (size_t)b * S * ROWSTR + (size_t)h * HD;
  const _Float16* qbase = Q + bh + (size_t)qr0 * ROWSTR;
  const _Float16* kbase = Km + bh;
  const _Float16* vbase = Vm + bh;

  // Q fragments resident in registers across the whole KV loop (16x128).
  v16h qf[4];
#pragma unroll
  for (int kf = 0; kf < 4; ++kf) qf[kf] = frag_a(qbase, ROWSTR, kf * 32);

  v8f o[8];
#pragma unroll
  for (int i = 0; i < 8; ++i) { v8f z = {}; o[i] = z; }
  float mrow[8], lrow[8];
#pragma unroll
  for (int i = 0; i < 8; ++i) { mrow[i] = -3.0e38f; lrow[i] = 0.f; }

  const int rsel = (lane >> 4) * 8;
  const int csel = lane & 15;
  _Float16* sPw = &sP[wave * 16 * KB];
  const int ldr = tid >> 2;         // kv row 0..63
  const int ldc = (tid & 3) * 32;   // hd offset

  for (int kb = 0; kb < S; kb += KB) {
    // --- K tile via async DMA ---------------------------------------------
    const _Float16* ks = kbase + (size_t)(kb + ldr) * ROWSTR + ldc;
#pragma unroll
    for (int c = 0; c < 4; ++c)
      async_ld16(lds_off(&sK[ldr * HD + ldc + c * 8]), ks + c * 8);

    // --- V tile transposed through VGPRs ----------------------------------
    const _Float16* vs = vbase + (size_t)(kb + ldr) * ROWSTR + ldc;
#pragma unroll
    for (int c = 0; c < 4; ++c) {
      v8h vv = *(const v8h*)(vs + c * 8);
#pragma unroll
      for (int j = 0; j < 8; ++j) sVt[(ldc + c * 8 + j) * KB + ldr] = vv[j];
    }
    if (kb + KB < S) {  // gfx1250 global_prefetch_b8 on next KV slab
      __builtin_prefetch(ks + (size_t)KB * ROWSTR, 0, 1);
      __builtin_prefetch(vs + (size_t)KB * ROWSTR, 0, 1);
    }
    wait_async0();
    __syncthreads();

    // --- S = Q K^T  (16 x 64 per wave) ------------------------------------
    v8f st[4];
#pragma unroll
    for (int nt = 0; nt < 4; ++nt) {
      v8f a = {};
#pragma unroll
      for (int kf = 0; kf < 4; ++kf)
        a = wmma32(qf[kf], frag_b(&sK[(nt * 16) * HD], HD, kf * 32), a);
      st[nt] = a;
    }

    // --- additive mask bias -----------------------------------------------
#pragma unroll
    for (int nt = 0; nt < 4; ++nt)
#pragma unroll
      for (int i = 0; i < 8; ++i) {
        const int r = qr0 + i + rsel;
        const int c = kb + nt * 16 + csel;
        const int mv = mask[((size_t)b * S + r) * S + c];
        st[nt][i] += (mv > 0) ? 0.f : -1.0e10f;
      }

    // --- streaming softmax (row stats via width-16 xor shuffles) ----------
    float pm[8];
#pragma unroll
    for (int i = 0; i < 8; ++i) {
      float x = st[0][i];
#pragma unroll
      for (int nt = 1; nt < 4; ++nt) x = fmaxf(x, st[nt][i]);
      x = fmaxf(x, __shfl_xor(x, 1, 16));
      x = fmaxf(x, __shfl_xor(x, 2, 16));
      x = fmaxf(x, __shfl_xor(x, 4, 16));
      x = fmaxf(x, __shfl_xor(x, 8, 16));
      pm[i] = x;
    }
    float alpha[8];
#pragma unroll
    for (int i = 0; i < 8; ++i) {
      const float mn = fmaxf(mrow[i], pm[i]);
      alpha[i] = __expf(mrow[i] - mn);
      mrow[i] = mn;
    }
    float psum[8];
#pragma unroll
    for (int i = 0; i < 8; ++i) psum[i] = 0.f;
#pragma unroll
    for (int nt = 0; nt < 4; ++nt)
#pragma unroll
      for (int i = 0; i < 8; ++i) {
        const float p = __expf(st[nt][i] - mrow[i]);
        st[nt][i] = p;
        psum[i] += p;
      }
#pragma unroll
    for (int i = 0; i < 8; ++i) {
      float s = psum[i];
      s += __shfl_xor(s, 1, 16);
      s += __shfl_xor(s, 2, 16);
      s += __shfl_xor(s, 4, 16);
      s += __shfl_xor(s, 8, 16);
      lrow[i] = lrow[i] * alpha[i] + s;
    }
#pragma unroll
    for (int ot = 0; ot < 8; ++ot)
#pragma unroll
      for (int i = 0; i < 8; ++i) o[ot][i] *= alpha[i];

    // --- stage P (f16) in per-wave LDS so it re-enters as A fragments -----
#pragma unroll
    for (int nt = 0; nt < 4; ++nt)
#pragma unroll
      for (int i = 0; i < 8; ++i)
        sPw[(i + rsel) * KB + nt * 16 + csel] = (_Float16)st[nt][i];

    // --- O += P @ V  (K-dim = 64 kv positions -> 2 WMMA steps) ------------
#pragma unroll
    for (int kf = 0; kf < 2; ++kf) {
      const v16h pa = frag_a(sPw, KB, kf * 32);
#pragma unroll
      for (int ot = 0; ot < 8; ++ot)
        o[ot] = wmma32(pa, frag_b(&sVt[(ot * 16) * KB], KB, kf * 32), o[ot]);
    }
    __syncthreads();
  }

  // --- normalize and write x[b, q, h, :] ----------------------------------
  _Float16* xout = X + bh + (size_t)qr0 * ROWSTR;
#pragma unroll
  for (int i = 0; i < 8; ++i) {
    const float inv = 1.0f / lrow[i];
#pragma unroll
    for (int ot = 0; ot < 8; ++ot)
      xout[(size_t)(i + rsel) * ROWSTR + ot * 16 + csel] =
          (_Float16)(o[ot][i] * inv);
  }
}

// ---------------------------------------------------------------------------
// Host-side orchestration
// ---------------------------------------------------------------------------
extern "C" void kernel_launch(void* const* d_in, const int* in_sizes, int n_in,
                              void* d_out, int out_size, void* d_ws,
                              size_t ws_size, hipStream_t stream) {
  (void)in_sizes; (void)n_in; (void)out_size; (void)ws_size;
  constexpr int B = 4, S = 2048, D = 2048;
  constexpr int M = B * S;   // 8192
  constexpr int N = 2048;    // H*HD

  const float* inq  = (const float*)d_in[0];
  const float* inkv = (const float*)d_in[1];
  const int*   mask = (const int*)d_in[2];
  const float* Wq = (const float*)d_in[3];
  const float* Wk = (const float*)d_in[4];
  const float* Wv = (const float*)d_in[5];
  const float* Wo = (const float*)d_in[6];
  float* out = (float*)d_out;

  char* ws = (char*)d_ws;
  size_t off = 0;
  auto alloc = [&](size_t elems) {
    _Float16* p = (_Float16*)(ws + off);
    off += elems * sizeof(_Float16);
    return p;
  };
  _Float16* inq_h  = alloc((size_t)M * D);
  _Float16* inkv_h = alloc((size_t)M * D);
  _Float16* wq_t   = alloc((size_t)D * N);
  _Float16* wk_t   = alloc((size_t)D * N);
  _Float16* wv_t   = alloc((size_t)D * N);
  _Float16* wo_t   = alloc((size_t)N * D);
  _Float16* q_h    = alloc((size_t)M * N);
  _Float16* k_h    = alloc((size_t)M * N);
  _Float16* v_h    = alloc((size_t)M * N);
  _Float16* x_h    = alloc((size_t)M * N);

  // 1) activation conversion
  {
    const int n4 = (M * D) / 4;
    dim3 g((n4 + 255) / 256);
    cvt4_f32_f16<<<g, 256, 0, stream>>>(inq,  inq_h,  n4);
    cvt4_f32_f16<<<g, 256, 0, stream>>>(inkv, inkv_h, n4);
  }
  // 2) weight transpose-convert: [K][N] f32 -> [N][K] f16
  {
    dim3 g(N / 32, D / 32), blk(32, 8);
    transpose_cvt<<<g, blk, 0, stream>>>(Wq, wq_t, D, N);
    transpose_cvt<<<g, blk, 0, stream>>>(Wk, wk_t, D, N);
    transpose_cvt<<<g, blk, 0, stream>>>(Wv, wv_t, D, N);
    transpose_cvt<<<g, blk, 0, stream>>>(Wo, wo_t, N, D);
  }
  // 3) QKV projections (Q pre-scaled by 1/sqrt(HD))
  {
    const float qscale = 0.08838834764831845f;  // 1/sqrt(128)
    dim3 gg(N / 128, M / 128);
    gemm_h<false><<<gg, 256, 0, stream>>>(inq_h,  wq_t, (void*)q_h, M, N, D, qscale);
    gemm_h<false><<<gg, 256, 0, stream>>>(inkv_h, wk_t, (void*)k_h, M, N, D, 1.0f);
    gemm_h<false><<<gg, 256, 0, stream>>>(inkv_h, wv_t, (void*)v_h, M, N, D, 1.0f);
  }
  // 4) flash attention
  attn_h<<<dim3(S / 128, B * 16), 256, 0, stream>>>(q_h, k_h, v_h, mask, x_h);
  // 5) output projection -> f32
  {
    dim3 go(D / 128, M / 128);
    gemm_h<true><<<go, 256, 0, stream>>>(x_h, wo_t, (void*)out, M, D, N, 1.0f);
  }
}
